// EncoderBlock_21526376088286
// MI455X (gfx1250) — matrix operand order repaired and checked by reading, not compiled, not run on hardware
//
#include <hip/hip_runtime.h>
#include <hip/hip_bf16.h>
#include <cstdint>
#include <cstddef>

// ---------------- problem constants ----------------
#define B_    16
#define L_    1536
#define E_    1024
#define F_    4096
#define TOPK  7          // int(1.0 * log(1536)) == 7
#define LN_EPS 1e-5f

// ---------------- WMMA tile config ----------------
#define BM    128
#define BN    128
#define BKK   32
#define LDST  40   // LDS row stride in bf16 elems (32 + 8 pad, keeps 16B alignment)

typedef unsigned short u16;
typedef __attribute__((ext_vector_type(16))) __bf16 v16bf;
typedef __attribute__((ext_vector_type(8)))  float  v8f;

__device__ __forceinline__ u16 f2bf(float f) {
  unsigned int u = __float_as_uint(f);
  u += 0x7FFFu + ((u >> 16) & 1u);          // round-to-nearest-even
  return (u16)(u >> 16);
}

union FragU { uint4 q[2]; v16bf v; };

// A fragment (16x32 bf16, M x K): lanes 0-15 row M=lane hold K 0..7 & 16..23,
// lanes 16-31 row M=lane-16 hold K 8..15 & 24..31  (ISA 7.12.2)
__device__ __forceinline__ v16bf frag_a(const u16* lds, int rowbase, int lane) {
  int m  = rowbase + (lane & 15);
  int kb = (lane & 16) ? 8 : 0;
  const uint4* p = (const uint4*)(lds + m * LDST + kb);
  FragU u; u.q[0] = p[0]; u.q[1] = p[2];    // +16 bf16 == +2 uint4
  return u.v;
}
// B fragment (32x16 bf16, K x N): lanes 0-15 col N=lane hold K 0..7 & 16..23,
// lanes 16-31 col N=lane-16 hold K 8..15 & 24..31
__device__ __forceinline__ v16bf frag_b(const u16* lds, int colbase, int lane) {
  int n  = colbase + (lane & 15);
  int kb = (lane & 16) ? 8 : 0;
  const uint4* p = (const uint4*)(lds + n * LDST + kb);
  FragU u; u.q[0] = p[0]; u.q[1] = p[2];
  return u.v;
}

__device__ __forceinline__ v8f wmma_bf16(v16bf a, v16bf b, v8f c) {
  return __builtin_amdgcn_wmma_f32_16x16x32_bf16(false, a, false, b, (short)0, c, false, false);
}

// async-stage a 128x32 bf16 tile (8KB) into LDS: one b128 per lane per 64-row slab.
// Issues exactly 2 ASYNCcnt-tracked ops per call.
__device__ __forceinline__ void stage_async(const u16* __restrict__ g, int ldg,
                                            unsigned int lds_base, int tid) {
  int r = tid >> 2;              // 0..63
  int c = (tid & 3) << 3;        // 0,8,16,24 (bf16 elems; 16B chunks)
#pragma unroll
  for (int rr = 0; rr < BM; rr += 64) {
    const void* src = (const void*)(g + (size_t)(r + rr) * ldg + c);
    unsigned int dst = lds_base + (unsigned int)(((r + rr) * LDST + c) * 2);
    asm volatile("global_load_async_to_lds_b128 %0, %1, off"
                 :: "v"(dst), "v"(src) : "memory");
  }
}

// ============ double-buffered async->LDS WMMA mainloop ============
// Stages slice ki+1 while computing slice ki. Async loads complete in order,
// so s_wait_asynccnt <= 4 (the 4 just-issued ops) implies slice ki has landed.
#define GEMM_MAINLOOP(Aptr, Bptr, KK)                                          \
  {                                                                            \
    const int nk = (KK) / BKK;                                                 \
    stage_async((Aptr), (KK), ldsA[0], tid);                                   \
    stage_async((Bptr), (KK), ldsB[0], tid);                                   \
    for (int ki = 0; ki < nk; ++ki) {                                          \
      const int cur = ki & 1;                                                  \
      if (ki + 1 < nk) {                                                       \
        stage_async((Aptr) + (size_t)(ki + 1) * BKK, (KK), ldsA[cur ^ 1], tid);\
        stage_async((Bptr) + (size_t)(ki + 1) * BKK, (KK), ldsB[cur ^ 1], tid);\
        asm volatile("s_wait_asynccnt 0x4" ::: "memory");                      \
      } else {                                                                 \
        asm volatile("s_wait_asynccnt 0x0" ::: "memory");                      \
      }                                                                        \
      if (ki + 2 < nk) {                 /* distance-2 L2 prefetch */          \
        int pr = tid >> 2, pc = (tid & 3) << 3;                                \
        __builtin_prefetch((Aptr) + (size_t)pr * (KK) + (ki + 2) * BKK + pc, 0, 0); \
        __builtin_prefetch((Bptr) + (size_t)pr * (KK) + (ki + 2) * BKK + pc, 0, 0); \
      }                                                                        \
      __syncthreads();                                                         \
      const u16* cA = &sA[cur][0];                                             \
      const u16* cB = &sB[cur][0];                                             \
      v16bf af[2], bfr[4];                                                     \
      _Pragma("unroll")                                                        \
      for (int mi = 0; mi < 2; ++mi) af[mi]  = frag_a(cA, wm * 32 + mi * 16, lane); \
      _Pragma("unroll")                                                        \
      for (int ni = 0; ni < 4; ++ni) bfr[ni] = frag_b(cB, wn * 64 + ni * 16, lane); \
      _Pragma("unroll")                                                        \
      for (int mi = 0; mi < 2; ++mi)                                           \
        _Pragma("unroll")                                                      \
        for (int ni = 0; ni < 4; ++ni)                                         \
          acc[mi][ni] = wmma_bf16(af[mi], bfr[ni], acc[mi][ni]);               \
      __syncthreads();                                                         \
    }                                                                          \
  }

// ============ GEMM: C[M,N] = A[M,K] @ Bw[N,K]^T + bias ============
// A, Bw in bf16; optional f32 and/or bf16 outputs; optional exact GELU.
template <int EPI, bool OF32, bool OBF>
__global__ void __launch_bounds__(256)
gemm_bf16_nt(const u16* __restrict__ A, const u16* __restrict__ Bw,
             const float* __restrict__ bias,
             float* __restrict__ Cf, u16* __restrict__ Cb,
             int N, int K) {
  __shared__ u16 sA[2][BM * LDST];
  __shared__ u16 sB[2][BN * LDST];
  const unsigned int ldsA[2] = {(unsigned int)(uintptr_t)&sA[0][0],
                                (unsigned int)(uintptr_t)&sA[1][0]};
  const unsigned int ldsB[2] = {(unsigned int)(uintptr_t)&sB[0][0],
                                (unsigned int)(uintptr_t)&sB[1][0]};
  const int tid  = threadIdx.x;
  const int lane = tid & 31, wave = tid >> 5;
  const int wm = wave & 3, wn = wave >> 2;          // 4x2 wave grid -> 32x64 per wave
  const int row0 = blockIdx.y * BM, col0 = blockIdx.x * BN;

  const v8f vzero = {0, 0, 0, 0, 0, 0, 0, 0};
  v8f acc[2][4];
#pragma unroll
  for (int mi = 0; mi < 2; ++mi)
#pragma unroll
    for (int ni = 0; ni < 4; ++ni) acc[mi][ni] = vzero;

  const u16* Ap = A  + (size_t)row0 * K;
  const u16* Bp = Bw + (size_t)col0 * K;
  GEMM_MAINLOOP(Ap, Bp, K)

  // C/D layout: VGPR j -> M = j (+8 for lanes 16-31); N = lane&15
#pragma unroll
  for (int mi = 0; mi < 2; ++mi)
#pragma unroll
    for (int ni = 0; ni < 4; ++ni) {
      int col = col0 + wn * 64 + ni * 16 + (lane & 15);
      float bv = bias[col];
#pragma unroll
      for (int j = 0; j < 8; ++j) {
        int row = row0 + wm * 32 + mi * 16 + j + ((lane >> 4) << 3);
        float v = acc[mi][ni][j] + bv;
        if (EPI == 1) v = 0.5f * v * (1.0f + erff(v * 0.70710678118654752f));
        if (OF32) Cf[(size_t)row * N + col] = v;
        if (OBF)  Cb[(size_t)row * N + col] = f2bf(v);
      }
    }
}

// ============ Gram-diagonal autocorrelation: mv[b,tau] += sum_t q[t].k[t-tau] ============
__global__ void __launch_bounds__(256)
corr_kernel(const u16* __restrict__ q, const u16* __restrict__ k,
            float* __restrict__ mv) {
  __shared__ u16 sA[2][BM * LDST];
  __shared__ u16 sB[2][BN * LDST];
  const unsigned int ldsA[2] = {(unsigned int)(uintptr_t)&sA[0][0],
                                (unsigned int)(uintptr_t)&sA[1][0]};
  const unsigned int ldsB[2] = {(unsigned int)(uintptr_t)&sB[0][0],
                                (unsigned int)(uintptr_t)&sB[1][0]};
  const int tid  = threadIdx.x;
  const int lane = tid & 31, wave = tid >> 5;
  const int wm = wave & 3, wn = wave >> 2;
  const int bb = blockIdx.z;
  const int row0 = blockIdx.y * BM;   // t
  const int col0 = blockIdx.x * BN;   // s

  const v8f vzero = {0, 0, 0, 0, 0, 0, 0, 0};
  v8f acc[2][4];
#pragma unroll
  for (int mi = 0; mi < 2; ++mi)
#pragma unroll
    for (int ni = 0; ni < 4; ++ni) acc[mi][ni] = vzero;

  const u16* Ap = q + (size_t)bb * L_ * E_ + (size_t)row0 * E_;
  const u16* Bp = k + (size_t)bb * L_ * E_ + (size_t)col0 * E_;
  GEMM_MAINLOOP(Ap, Bp, E_)

  float* mvb = mv + (size_t)bb * L_;
#pragma unroll
  for (int mi = 0; mi < 2; ++mi)
#pragma unroll
    for (int ni = 0; ni < 4; ++ni) {
      int s = col0 + wn * 64 + ni * 16 + (lane & 15);
#pragma unroll
      for (int j = 0; j < 8; ++j) {
        int t = row0 + wm * 32 + mi * 16 + j + ((lane >> 4) << 3);
        int tau = t - s; if (tau < 0) tau += L_;
        atomicAdd(mvb + tau, acc[mi][ni][j]);
      }
    }
}

// ============ small kernels ============
__global__ void zero_kernel(float* p, int n) {
  int i = blockIdx.x * 256 + threadIdx.x;
  if (i < n) p[i] = 0.0f;
}

__global__ void cvt_bf16_kernel(const float* __restrict__ in, u16* __restrict__ outp,
                                size_t n) {
  size_t i = (size_t)blockIdx.x * 256 + threadIdx.x;
  if (i < n) outp[i] = f2bf(in[i]);
}

// top-7 over batch-mean of mv, then per-batch softmax of the picked lags
__global__ void __launch_bounds__(256)
topk_softmax_kernel(const float* __restrict__ mv, int* __restrict__ idx,
                    float* __restrict__ wts) {
  __shared__ float s_avg[L_];
  __shared__ float s_val[256];
  __shared__ int   s_ind[256];
  __shared__ int   s_top[TOPK];
  const int tid = threadIdx.x;
  for (int t = tid; t < L_; t += 256) {
    float s = 0.0f;
    for (int b = 0; b < B_; ++b) s += mv[(size_t)b * L_ + t];
    s_avg[t] = s;
  }
  __syncthreads();
  for (int i = 0; i < TOPK; ++i) {
    float best = -3.4e38f; int bi = 0;
    for (int t = tid; t < L_; t += 256)
      if (s_avg[t] > best) { best = s_avg[t]; bi = t; }
    s_val[tid] = best; s_ind[tid] = bi;
    __syncthreads();
    for (int off = 128; off > 0; off >>= 1) {
      if (tid < off && s_val[tid + off] > s_val[tid]) {
        s_val[tid] = s_val[tid + off]; s_ind[tid] = s_ind[tid + off];
      }
      __syncthreads();
    }
    if (tid == 0) { s_top[i] = s_ind[0]; idx[i] = s_ind[0]; s_avg[s_ind[0]] = -3.4e38f; }
    __syncthreads();
  }
  if (tid < B_) {
    const float scale = 1.0f / ((float)E_ * (float)L_);  // irfft 1/L and mean over E
    float w[TOPK]; float m = -3.4e38f;
    for (int i = 0; i < TOPK; ++i) {
      w[i] = mv[(size_t)tid * L_ + s_top[i]] * scale;
      m = fmaxf(m, w[i]);
    }
    float ssum = 0.0f;
    for (int i = 0; i < TOPK; ++i) { w[i] = __expf(w[i] - m); ssum += w[i]; }
    for (int i = 0; i < TOPK; ++i) wts[tid * TOPK + i] = w[i] / ssum;
  }
}

// agg[b,t,e] = sum_i wts[b,i] * v[b,(t+idx[i])%L,e]  -> bf16 (feeds O-proj GEMM)
__global__ void __launch_bounds__(256)
agg_kernel(const float* __restrict__ v, const int* __restrict__ idx,
           const float* __restrict__ wts, u16* __restrict__ outb) {
  size_t i = (size_t)blockIdx.x * 256 + threadIdx.x;   // over B*L*E
  int e = (int)(i % E_);
  size_t bt = i / E_;
  int t = (int)(bt % L_);
  int b = (int)(bt / L_);
  const float* vb = v + ((size_t)b * L_) * E_;
  float s = 0.0f;
#pragma unroll
  for (int j = 0; j < TOPK; ++j) {
    int t2 = t + idx[j]; if (t2 >= L_) t2 -= L_;
    s += wts[b * TOPK + j] * vb[(size_t)t2 * E_ + e];
  }
  outb[i] = f2bf(s);
}

// out = (a+b) - movavg25_edgepad(a+b); optional bf16 mirror for the next GEMM
template <bool WBF>
__global__ void __launch_bounds__(256)
resid_decomp_kernel(const float* __restrict__ a, const float* __restrict__ b,
                    float* __restrict__ outf, u16* __restrict__ outb) {
  size_t i = (size_t)blockIdx.x * 256 + threadIdx.x;   // over B*L*E
  int e = (int)(i % E_);
  size_t bt = i / E_;
  int t  = (int)(bt % L_);
  int bb = (int)(bt / L_);
  const float* pa = a + ((size_t)bb * L_) * E_ + e;
  const float* pb = b + ((size_t)bb * L_) * E_ + e;
  float sum = 0.0f;
#pragma unroll
  for (int j = -12; j <= 12; ++j) {
    int tc = t + j;
    tc = tc < 0 ? 0 : (tc > L_ - 1 ? L_ - 1 : tc);
    sum += pa[(size_t)tc * E_] + pb[(size_t)tc * E_];
  }
  float y = pa[(size_t)t * E_] + pb[(size_t)t * E_];
  float r = y - sum * (1.0f / 25.0f);
  outf[i] = r;
  if (WBF) outb[i] = f2bf(r);
}

// LayerNorm over E
__global__ void __launch_bounds__(256)
ln_kernel(const float* __restrict__ x2, const float* __restrict__ gamma,
          const float* __restrict__ beta, float* __restrict__ out) {
  __shared__ float rs[256], rq[256];
  const int row = blockIdx.x;                          // b*L + t
  const float* p = x2 + (size_t)row * E_;
  float s = 0.0f, sq = 0.0f;
  for (int e = threadIdx.x; e < E_; e += 256) { float v = p[e]; s += v; sq += v * v; }
  rs[threadIdx.x] = s; rq[threadIdx.x] = sq;
  __syncthreads();
  for (int off = 128; off > 0; off >>= 1) {
    if (threadIdx.x < off) {
      rs[threadIdx.x] += rs[threadIdx.x + off];
      rq[threadIdx.x] += rq[threadIdx.x + off];
    }
    __syncthreads();
  }
  float mu  = rs[0] * (1.0f / E_);
  float var = rq[0] * (1.0f / E_) - mu * mu;
  float rstd = rsqrtf(var + LN_EPS);
  for (int e = threadIdx.x; e < E_; e += 256)
    out[(size_t)row * E_ + e] = (p[e] - mu) * rstd * gamma[e] + beta[e];
}

// out[b,t,e] = xh[b,t,e] - mean_t xh[b,:,e]
__global__ void __launch_bounds__(256)
tmean_sub_kernel(const float* __restrict__ xh, float* __restrict__ out) {
  int i = blockIdx.x * 256 + threadIdx.x;              // over B*E
  int e = i % E_, b = i / E_;
  const float* p = xh + ((size_t)b * L_) * E_ + e;
  float s = 0.0f;
  for (int t = 0; t < L_; ++t) s += p[(size_t)t * E_];
  float m = s * (1.0f / L_);
  float* o = out + ((size_t)b * L_) * E_ + e;
  for (int t = 0; t < L_; ++t) o[(size_t)t * E_] = p[(size_t)t * E_] - m;
}

// ================================ driver ================================
extern "C" void kernel_launch(void* const* d_in, const int* in_sizes, int n_in,
                              void* d_out, int out_size, void* d_ws, size_t ws_size,
                              hipStream_t stream) {
  const float* x  = (const float*)d_in[0];
  const float* Wq = (const float*)d_in[1];  const float* bq = (const float*)d_in[2];
  const float* Wk = (const float*)d_in[3];  const float* bk = (const float*)d_in[4];
  const float* Wv = (const float*)d_in[5];  const float* bv = (const float*)d_in[6];
  const float* Wo = (const float*)d_in[7];  const float* bo = (const float*)d_in[8];
  const float* W1 = (const float*)d_in[9];  const float* b1 = (const float*)d_in[10];
  const float* W2 = (const float*)d_in[11]; const float* b2 = (const float*)d_in[12];
  const float* gamma = (const float*)d_in[13];
  const float* beta  = (const float*)d_in[14];
  float* out = (float*)d_out;

  const size_t BLE = (size_t)B_ * L_ * E_;        // 25,165,824
  const size_t BLF = (size_t)B_ * L_ * F_;        // 100,663,296
  const size_t EE  = (size_t)E_ * E_;
  const size_t EF  = (size_t)E_ * F_;

  char* base = (char*)d_ws;
  // f32 region
  float* f0 = (float*)base;                       // v  -> ffn2 out
  float* f1 = f0 + BLE;                           // attn -> x2
  float* f2 = f1 + BLE;                           // x1 -> xh
  float* mv = f2 + BLE;                           // B*L lag bins
  float* wt = mv + (size_t)B_ * L_;               // B*TOPK weights
  int*   ix = (int*)(wt + (size_t)B_ * TOPK);     // TOPK lag indices
  // bf16 region (1MB after small buffers for alignment slack)
  u16* bfA = (u16*)(base + 3 * BLE * 4 + (1 << 20));  // xbf -> aggbf -> x1bf
  u16* bfQ = bfA + BLE;
  u16* bfK = bfQ + BLE;
  u16* bfH = bfK + BLE;                           // gelu hidden, BLF elems
  u16* wqb = bfH + BLF;
  u16* wkb = wqb + EE;
  u16* wvb = wkb + EE;
  u16* wob = wvb + EE;
  u16* w1b = wob + EE;                            // F x E
  u16* w2b = w1b + EF;                            // E x F

  const int ML = B_ * L_;                         // 24576 rows
  dim3 blk(256);
  dim3 gproj(E_ / BN, ML / BM);                   // (8, 192)
  dim3 gffn1(F_ / BN, ML / BM);                   // (32, 192)
  dim3 gcorr(L_ / BN, L_ / BM, B_);               // (12, 12, 16)
  const unsigned gBLE = (unsigned)(BLE / 256);

  // 0) one-time bf16 conversions (x + all weights)
  cvt_bf16_kernel<<<gBLE, blk, 0, stream>>>(x, bfA, BLE);
  cvt_bf16_kernel<<<(unsigned)(EE / 256), blk, 0, stream>>>(Wq, wqb, EE);
  cvt_bf16_kernel<<<(unsigned)(EE / 256), blk, 0, stream>>>(Wk, wkb, EE);
  cvt_bf16_kernel<<<(unsigned)(EE / 256), blk, 0, stream>>>(Wv, wvb, EE);
  cvt_bf16_kernel<<<(unsigned)(EE / 256), blk, 0, stream>>>(Wo, wob, EE);
  cvt_bf16_kernel<<<(unsigned)(EF / 256), blk, 0, stream>>>(W1, w1b, EF);
  cvt_bf16_kernel<<<(unsigned)(EF / 256), blk, 0, stream>>>(W2, w2b, EF);

  // 1) Q, K (bf16-only outputs; only consumed by corr), V (f32; consumed by agg)
  gemm_bf16_nt<0, false, true ><<<gproj, blk, 0, stream>>>(bfA, wqb, bq, (float*)nullptr, bfQ, E_, E_);
  gemm_bf16_nt<0, false, true ><<<gproj, blk, 0, stream>>>(bfA, wkb, bk, (float*)nullptr, bfK, E_, E_);
  gemm_bf16_nt<0, true,  false><<<gproj, blk, 0, stream>>>(bfA, wvb, bv, f0, (u16*)nullptr, E_, E_);

  // 2) lag correlation via Gram diagonals
  zero_kernel<<<(B_ * L_ + 255) / 256, blk, 0, stream>>>(mv, B_ * L_);
  corr_kernel<<<gcorr, blk, 0, stream>>>(bfQ, bfK, mv);

  // 3) top-k lags + per-batch softmax weights
  topk_softmax_kernel<<<1, blk, 0, stream>>>(mv, ix, wt);

  // 4) time-delay aggregation of V -> bf16 (reuse bfA; xbf is dead)
  agg_kernel<<<gBLE, blk, 0, stream>>>(f0, ix, wt, bfA);

  // 5) output projection -> attn f32 (f1)
  gemm_bf16_nt<0, true, false><<<gproj, blk, 0, stream>>>(bfA, wob, bo, f1, (u16*)nullptr, E_, E_);

  // 6) x1 = (x + attn) - movavg -> f32 (f2) + bf16 (bfA, agg is dead)
  resid_decomp_kernel<true><<<gBLE, blk, 0, stream>>>(x, f1, f2, bfA);

  // 7) h = gelu(x1 @ W1^T + b1) -> bf16 only (halves h traffic vs f32)
  gemm_bf16_nt<1, false, true><<<gffn1, blk, 0, stream>>>(bfA, w1b, b1, (float*)nullptr, bfH, F_, E_);

  // 8) f = h @ W2^T + b2 -> f32 (f0; v is dead)
  gemm_bf16_nt<0, true, false><<<gproj, blk, 0, stream>>>(bfH, w2b, b2, f0, (u16*)nullptr, E_, F_);

  // 9) x2 = (x1 + f) - movavg -> f32 (f1; attn is dead)
  resid_decomp_kernel<false><<<gBLE, blk, 0, stream>>>(f2, f0, f1, (u16*)nullptr);

  // 10) LayerNorm -> xh (f2; x1 is dead)
  ln_kernel<<<ML, blk, 0, stream>>>(f1, gamma, beta, f2);

  // 11) subtract per-(b,e) time mean -> output
  tmean_sub_kernel<<<(B_ * E_) / 256, blk, 0, stream>>>(f2, out);
}
